// EfficientAttention_78254304133357
// MI455X (gfx1250) — matrix operand order repaired and checked by reading, not compiled, not run on hardware
//
#include <hip/hip_runtime.h>
#include <cstdint>

// ---------------------------------------------------------------------------
// EfficientAttention on MI455X (gfx1250, wave32, WMMA bf16, async-LDS staging)
//
//   h  = x*w_in + b_in                       (rank-1, elementwise, bf16-packed)
//   Q/K/V = h @ w{q,k,v}                     (WMMA bf16 GEMM, f32 accum,
//                                             double-buffered async LDS tiles)
//   attn  = softmax(causal(QK^T/8)) @ V      (flash-style, WMMA, S x S never
//                                             materialized)
//   out   = attn_out . (wo @ wf) + (bo.wf + bf)   (linear tail folded)
//
// All B-operands (weights, V) are stored pair-packed and *pair-minor* so every
// WMMA fragment load is a contiguous b128 (no register shuffles).
// ---------------------------------------------------------------------------

#define DIMN   1024
#define HEADS  16
#define HD     64
#define BB     2
#define SS     2048
#define MTOT   (BB * SS)          // 4096 rows
#define KP     (DIMN / 2)         // 512 K-pairs per GEMM
#define SP     (SS / 2)           // 1024 key-pairs per batch

typedef __attribute__((ext_vector_type(16))) __bf16   v16bf;
typedef __attribute__((ext_vector_type(8)))  float    v8f;
typedef __attribute__((ext_vector_type(8)))  unsigned v8u;
typedef int v4i_ __attribute__((vector_size(16)));

union F16 { v8u u; v16bf b; };    // 32B fragment reinterpret

__device__ __forceinline__ __bf16 f2bf(float f) {
    unsigned u = __builtin_bit_cast(unsigned, f);
    unsigned r = u + 0x7FFFu + ((u >> 16) & 1u);      // round-to-nearest-even
    unsigned short h = (unsigned short)(r >> 16);
    return __builtin_bit_cast(__bf16, h);
}
__device__ __forceinline__ unsigned packbf2(float lo, float hi) {
    unsigned l = __builtin_bit_cast(unsigned short, f2bf(lo));
    unsigned h = __builtin_bit_cast(unsigned short, f2bf(hi));
    return l | (h << 16);
}

__device__ __forceinline__ v8f wmma_bf16(v16bf a, v16bf b, v8f c) {
    return __builtin_amdgcn_wmma_f32_16x16x32_bf16(
        false, a, false, b, (short)0, c, false, false);
}

// ---- CDNA5 async global->LDS copy (ASYNCcnt path), with sync fallback ------
#define AS1 __attribute__((address_space(1)))
#define AS3 __attribute__((address_space(3)))
#if defined(__has_builtin)
#if __has_builtin(__builtin_amdgcn_global_load_async_to_lds_b128)
#define HAVE_ASYNC_LDS 1
#endif
#endif

__device__ __forceinline__ void async_cp16(void* lds_dst, const void* gsrc) {
#if defined(HAVE_ASYNC_LDS)
    __builtin_amdgcn_global_load_async_to_lds_b128(
        (AS1 v4i_*)(uintptr_t)gsrc,
        (AS3 v4i_*)(unsigned)(uintptr_t)lds_dst, 0, 0);
#else
    *(uint4*)lds_dst = *(const uint4*)gsrc;
#endif
}

#if defined(HAVE_ASYNC_LDS)
#if __has_builtin(__builtin_amdgcn_s_wait_asynccnt)
#define ASYNC_WAIT(n) __builtin_amdgcn_s_wait_asynccnt(n)
#else
#define ASYNC_WAIT(n) asm volatile("s_wait_asynccnt %0" ::"i"(n) : "memory")
#endif
#else
#define ASYNC_WAIT(n) ((void)0)
#endif

// ---------------------------------------------------------------------------
// Kernel 1: fp32 -> bf16 convert of wq|wk|wv, K-pair packed, PAIR-MINOR:
//   Wt[z][n][p] (uint) = { bf16(W[2p][n]), bf16(W[2p+1][n]) }
//   (reads coalesced along n; scattered writes absorbed by L2)
// ---------------------------------------------------------------------------
__global__ void cvt_weights_kernel(const float* __restrict__ wq,
                                   const float* __restrict__ wk,
                                   const float* __restrict__ wv,
                                   unsigned* __restrict__ out) {
    unsigned i = blockIdx.x * blockDim.x + threadIdx.x;  // 3*KP*DIMN
    if (i >= 3u * KP * DIMN) return;
    unsigned z = i / (KP * DIMN);
    unsigned rem = i - z * (KP * DIMN);
    unsigned p = rem / DIMN, n = rem - p * DIMN;         // n fastest -> reads ok
    const float* src = (z == 0) ? wq : (z == 1) ? wk : wv;
    out[(size_t)z * (DIMN * KP) + (size_t)n * KP + p] =
        packbf2(src[(size_t)(2 * p) * DIMN + n],
                src[(size_t)(2 * p + 1) * DIMN + n]);
}

// ---------------------------------------------------------------------------
// Kernel 2: h[m][d] = x[m]*w_in[d] + b_in[d], bf16, written as uint pairs
// ---------------------------------------------------------------------------
__global__ void make_h_kernel(const float* __restrict__ x,
                              const float* __restrict__ w_in,
                              const float* __restrict__ b_in,
                              unsigned* __restrict__ hu) {
    unsigned i = blockIdx.x * blockDim.x + threadIdx.x;  // MTOT*DIMN/2
    if (i >= (unsigned)MTOT * DIMN / 2) return;
    unsigned m = i >> 9, d2 = (i & 511) * 2;
    float xv = x[m];
    hu[i] = packbf2(xv * w_in[d2] + b_in[d2],
                    xv * w_in[d2 + 1] + b_in[d2 + 1]);
}

// ---------------------------------------------------------------------------
// Kernel 3: fold linear tail:  wof[d] = sum_k wo[d][k]*wf[k];
//           wof[DIMN] = bo.wf + bf
// ---------------------------------------------------------------------------
__global__ void make_wof_kernel(const float* __restrict__ wo,
                                const float* __restrict__ bo,
                                const float* __restrict__ wf,
                                const float* __restrict__ bfin,
                                float* __restrict__ wof) {
    int d = blockIdx.x * blockDim.x + threadIdx.x;
    if (d < DIMN) {
        float s = 0.f;
        for (int k = 0; k < DIMN; ++k) s += wo[(size_t)d * DIMN + k] * wf[k];
        wof[d] = s;
    } else if (d == DIMN) {
        float s = 0.f;
        for (int k = 0; k < DIMN; ++k) s += bo[k] * wf[k];
        wof[DIMN] = s + bfin[0];
    }
}

// ---------------------------------------------------------------------------
// Kernel 4: QKV GEMM with double-buffered async-LDS staging (peeled epilogue).
//   grid = (MTOT/128, DIMN/64, 3), block = 256 (8 wave32)
//   block tile 128x64, wave tile 32x32 (2x2 WMMA), K step 32.
//   A frag: 2x ds_load_b128.  B frag: 2x ds_load_b128 (pair-minor LDS tile).
//   z==2 (V) stored d-major / key-pair-minor: Vt[b][d][k/2].
// ---------------------------------------------------------------------------
__global__ __launch_bounds__(256) void qkv_gemm_kernel(
        const __bf16* __restrict__ A,       // h [MTOT][DIMN] bf16
        const unsigned* __restrict__ Wt,    // packed weights [3][DIMN][KP]
        __bf16* __restrict__ Call) {        // Q | K | Vt(packed)
    __shared__ __bf16   As[2][128][40];     // 32 K-cols + pad (80B rows)
    __shared__ unsigned Bt[2][64][20];      // 64 N-cols x 16 K-pairs + pad

    const int z = blockIdx.z;
    const unsigned* W = Wt + (size_t)z * (DIMN * KP);

    const int m0 = blockIdx.x * 128;
    const int n0 = blockIdx.y * 64;
    const int tid = threadIdx.x;
    const int lane = tid & 31, wave = tid >> 5;
    const int wm = (wave >> 1) * 32;
    const int wn = (wave & 1) * 32;
    const int r = lane & 15, half = lane >> 4;

    auto stage = [&](int buf, int kk) {
        // A: 128 rows x 32 bf16 = 512 x 16B chunks, 2 per thread
        #pragma unroll
        for (int i = 0; i < 2; ++i) {
            int c = tid + i * 256;
            int row = c >> 2, ch = c & 3;
            async_cp16(&As[buf][row][ch * 8],
                       A + (size_t)(m0 + row) * DIMN + kk + ch * 8);
        }
        // B: 64 cols x 16 pairs = 256 x 16B chunks, 1 per thread (pair-minor)
        {
            int col = tid >> 2, ch = tid & 3;
            async_cp16(&Bt[buf][col][ch * 4],
                       W + (size_t)(n0 + col) * KP + kk / 2 + ch * 4);
        }
    };

    v8f acc[2][2] = {};

    auto compute = [&](int cur) {
        v16bf af[2], bfr[2];
        #pragma unroll
        for (int mi = 0; mi < 2; ++mi) {       // A frag: 2x b128 from LDS
            F16 f;
            const unsigned* ar = (const unsigned*)&As[cur][wm + mi * 16 + r][0];
            #pragma unroll
            for (int j = 0; j < 4; ++j) {
                f.u[j]     = ar[half * 8 + j];
                f.u[4 + j] = ar[half * 8 + 4 + j];
            }
            af[mi] = f.b;
        }
        #pragma unroll
        for (int ni = 0; ni < 2; ++ni) {       // B frag: 2x b128 (contiguous)
            F16 f;
            const unsigned* br = &Bt[cur][wn + ni * 16 + r][half * 8];
            #pragma unroll
            for (int v = 0; v < 8; ++v) f.u[v] = br[v];
            bfr[ni] = f.b;
        }
        #pragma unroll
        for (int mi = 0; mi < 2; ++mi)
            #pragma unroll
            for (int ni = 0; ni < 2; ++ni)
                acc[mi][ni] = wmma_bf16(af[mi], bfr[ni], acc[mi][ni]);
    };

    stage(0, 0);
    int kk = 0;
    for (; kk < DIMN - 32; kk += 32) {        // hot loop: unconditional stage
        const int cur = (kk >> 5) & 1;
        stage(cur ^ 1, kk + 32);
        ASYNC_WAIT(3);                        // newest 3 (next tile) may pend
        __syncthreads();
        compute(cur);
        __syncthreads();
    }
    ASYNC_WAIT(0);                            // peeled last tile
    __syncthreads();
    compute((kk >> 5) & 1);

    // store: C layout VGPR q -> row half*8+q, col r
    if (z < 2) {
        __bf16* C = Call + (size_t)z * MTOT * DIMN;
        #pragma unroll
        for (int mi = 0; mi < 2; ++mi)
            #pragma unroll
            for (int ni = 0; ni < 2; ++ni) {
                int rowb = m0 + wm + mi * 16 + half * 8;
                int colb = n0 + wn + ni * 16 + r;
                #pragma unroll
                for (int q = 0; q < 8; ++q)
                    C[(size_t)(rowb + q) * DIMN + colb] = f2bf(acc[mi][ni][q]);
            }
    } else {
        // Vt[b][d][k/2]: pack consecutive key rows, store key-pair-minor
        unsigned* Vt = (unsigned*)(Call + (size_t)2 * MTOT * DIMN);
        #pragma unroll
        for (int mi = 0; mi < 2; ++mi)
            #pragma unroll
            for (int ni = 0; ni < 2; ++ni) {
                int rowb = m0 + wm + mi * 16 + half * 8;   // even, global key
                int bb   = rowb >> 11;                     // /SS
                int srow = rowb & (SS - 1);
                int colb = n0 + wn + ni * 16 + r;          // d
                size_t basei = (size_t)bb * (DIMN * SP) + (size_t)colb * SP;
                #pragma unroll
                for (int q = 0; q < 8; q += 2)
                    Vt[basei + ((srow + q) >> 1)] =
                        packbf2(acc[mi][ni][q], acc[mi][ni][q + 1]);
            }
    }
}

// ---------------------------------------------------------------------------
// Kernel 5: fused causal flash attention + wof readout (per-head partials).
//   grid = (SS/128, BB*HEADS), block = 256; each wave owns a 16-row Q tile.
// ---------------------------------------------------------------------------
__global__ __launch_bounds__(256) void flash_attn_kernel(
        const __bf16* __restrict__ Q, const __bf16* __restrict__ K,
        const unsigned* __restrict__ Vt,     // [BB][DIMN][SP] key-pair-minor
        const float* __restrict__ wof, float* __restrict__ partial) {
    __shared__ __bf16 Pb[8][16][40];         // per-wave P staging (80B rows)

    const int bh = blockIdx.y, bb = bh / HEADS, h = bh % HEADS;
    const int tid = threadIdx.x, lane = tid & 31, wave = tid >> 5;
    const int q0 = blockIdx.x * 128 + wave * 16;
    const int r = lane & 15, half = lane >> 4;

    const size_t base = (size_t)bb * SS * DIMN + (size_t)h * HD;
    const __bf16* Qh = Q + base;
    const __bf16* Kh = K + base;
    const int dcol0 = h * HD;
    const size_t vbase = (size_t)bb * (DIMN * SP);

    // Q A-fragments (d 0..31, 32..63): 2x b128 each
    v16bf aQ[2];
    {
        const unsigned* qrow = (const unsigned*)(Qh + (size_t)(q0 + r) * DIMN);
        #pragma unroll
        for (int t = 0; t < 2; ++t) {
            F16 f;
            #pragma unroll
            for (int j = 0; j < 4; ++j) {
                f.u[j]     = qrow[t * 16 + half * 8 + j];
                f.u[4 + j] = qrow[t * 16 + half * 8 + 4 + j];
            }
            aQ[t] = f.b;
        }
    }

    float m_i[8], l_i[8], alpha[8];
    #pragma unroll
    for (int q = 0; q < 8; ++q) { m_i[q] = -1e30f; l_i[q] = 0.f; }
    v8f o[4] = {};

    const float scale = 0.125f;              // HD^-0.5
    const int qmax = q0 + 15;

    for (int kb = 0; kb <= qmax; kb += 32) {
        if (kb + 32 <= qmax) {               // prefetch next key block
            __builtin_prefetch(Kh + (size_t)(kb + 32 + r) * DIMN, 0, 0);
            __builtin_prefetch(Vt + vbase + (size_t)(dcol0 + r) * SP
                                  + ((kb + 32) >> 1), 0, 0);
        }

        // ---- scores: two 16-key sub-tiles, contraction over 64 dims ----
        v8f s[2];
        #pragma unroll
        for (int t = 0; t < 2; ++t) {
            const unsigned* krow =
                (const unsigned*)(Kh + (size_t)(kb + t * 16 + r) * DIMN);
            F16 f0, f1;                      // K^T frags, d 0..31 / 32..63
            #pragma unroll
            for (int v = 0; v < 8; ++v) {
                f0.u[v] = krow[half * 8 + v];
                f1.u[v] = krow[16 + half * 8 + v];
            }
            v8f zz = {};
            zz = wmma_bf16(aQ[0], f0.b, zz);
            zz = wmma_bf16(aQ[1], f1.b, zz);
            s[t] = zz;
        }

        // ---- online softmax update (row = half*8+q, col = r) ----
        float p0[8], p1[8];
        #pragma unroll
        for (int q = 0; q < 8; ++q) {
            int qi = q0 + half * 8 + q;
            float s0 = s[0][q] * scale, s1 = s[1][q] * scale;
            if (kb + r > qi)      s0 = -1e30f;   // causal mask
            if (kb + 16 + r > qi) s1 = -1e30f;
            float mx = fmaxf(s0, s1);
            #pragma unroll
            for (int d = 1; d < 16; d <<= 1) mx = fmaxf(mx, __shfl_xor(mx, d, 32));
            float mnew = fmaxf(m_i[q], mx);
            float al = __expf(m_i[q] - mnew);
            float e0 = __expf(s0 - mnew), e1 = __expf(s1 - mnew);
            float rs = e0 + e1;
            #pragma unroll
            for (int d = 1; d < 16; d <<= 1) rs += __shfl_xor(rs, d, 32);
            l_i[q] = l_i[q] * al + rs;
            m_i[q] = mnew;
            alpha[q] = al; p0[q] = e0; p1[q] = e1;
        }

        // ---- C-layout P -> A-layout via wave-private LDS slab ----
        #pragma unroll
        for (int q = 0; q < 8; ++q) {
            Pb[wave][half * 8 + q][r]      = f2bf(p0[q]);
            Pb[wave][half * 8 + q][16 + r] = f2bf(p1[q]);
        }
        F16 fp;
        {
            const unsigned* prow = (const unsigned*)&Pb[wave][r][0];
            #pragma unroll
            for (int j = 0; j < 4; ++j) {
                fp.u[j]     = prow[half * 8 + j];
                fp.u[4 + j] = prow[half * 8 + 4 + j];
            }
        }

        // ---- rescale O, accumulate P @ V over four 16-wide d chunks ----
        #pragma unroll
        for (int c = 0; c < 4; ++c) {
            #pragma unroll
            for (int q = 0; q < 8; ++q) o[c][q] *= alpha[q];
        }
        #pragma unroll
        for (int c = 0; c < 4; ++c) {
            // 32 keys x 16 dims, key-pair-minor: contiguous -> 2x b128
            F16 fv;
            const unsigned* vrow = Vt + vbase
                + (size_t)(dcol0 + c * 16 + r) * SP + (kb >> 1) + half * 8;
            #pragma unroll
            for (int v = 0; v < 8; ++v) fv.u[v] = vrow[v];
            o[c] = wmma_bf16(fp.b, fv.b, o[c]);
        }
    }

    // ---- epilogue: per-head partial = (O/l) . wof[head slice] ----
    float res[8];
    #pragma unroll
    for (int q = 0; q < 8; ++q) {
        float acc = 0.f;
        #pragma unroll
        for (int c = 0; c < 4; ++c) acc += o[c][q] * wof[dcol0 + c * 16 + r];
        #pragma unroll
        for (int d = 1; d < 16; d <<= 1) acc += __shfl_xor(acc, d, 32);
        res[q] = acc / l_i[q];
    }
    if (r == 0) {
        #pragma unroll
        for (int q = 0; q < 8; ++q) {
            int qi = q0 + half * 8 + q;
            partial[((size_t)bb * HEADS + h) * SS + qi] = res[q];
        }
    }
}

// ---------------------------------------------------------------------------
// Kernel 6: sum head partials + folded bias -> out [B,1,S]
// ---------------------------------------------------------------------------
__global__ void reduce_heads_kernel(const float* __restrict__ partial,
                                    const float* __restrict__ wof,
                                    float* __restrict__ out) {
    int i = blockIdx.x * blockDim.x + threadIdx.x;    // BB*SS
    if (i >= BB * SS) return;
    int b = i / SS, s = i % SS;
    float acc = wof[DIMN];
    #pragma unroll
    for (int h = 0; h < HEADS; ++h)
        acc += partial[((size_t)b * HEADS + h) * SS + s];
    out[(size_t)b * SS + s] = acc;
}

// ---------------------------------------------------------------------------
extern "C" void kernel_launch(void* const* d_in, const int* in_sizes, int n_in,
                              void* d_out, int out_size, void* d_ws, size_t ws_size,
                              hipStream_t stream) {
    const float* x    = (const float*)d_in[0];
    const float* w_in = (const float*)d_in[1];
    const float* b_in = (const float*)d_in[2];
    const float* wq   = (const float*)d_in[3];
    const float* wk   = (const float*)d_in[4];
    const float* wv   = (const float*)d_in[5];
    const float* wo   = (const float*)d_in[6];
    const float* bo   = (const float*)d_in[7];
    const float* wf   = (const float*)d_in[8];
    const float* bfin = (const float*)d_in[9];
    float* out = (float*)d_out;

    char* ws = (char*)d_ws;
    const size_t MB = 1024 * 1024;
    unsigned* h_u  = (unsigned*)(ws);               //  8 MB  h packed pairs
    unsigned* w_t  = (unsigned*)(ws + 8 * MB);      //  6 MB  packed pair-minor
    __bf16*   QKV  = (__bf16*)(ws + 14 * MB);       // 24 MB  Q | K | Vt(packed)
    float*    wofv = (float*)(ws + 38 * MB);        //  (DIMN+1) floats
    float*    prts = (float*)(ws + 38 * MB + 8192); // 256 KB [B][H][S]

    cvt_weights_kernel<<<(3u * KP * DIMN + 255) / 256, 256, 0, stream>>>(
        wq, wk, wv, w_t);
    make_h_kernel<<<((unsigned)MTOT * DIMN / 2 + 255) / 256, 256, 0, stream>>>(
        x, w_in, b_in, h_u);
    make_wof_kernel<<<(DIMN + 1 + 255) / 256, 256, 0, stream>>>(
        wo, bo, wf, bfin, wofv);

    dim3 gg(MTOT / 128, DIMN / 64, 3);
    qkv_gemm_kernel<<<gg, 256, 0, stream>>>((const __bf16*)h_u, w_t, QKV);

    const __bf16*   Qp = QKV;
    const __bf16*   Kp = QKV + (size_t)MTOT * DIMN;
    const unsigned* Vt = (const unsigned*)(QKV + (size_t)2 * MTOT * DIMN);
    dim3 gf(SS / 128, BB * HEADS);
    flash_attn_kernel<<<gf, 256, 0, stream>>>(Qp, Kp, Vt, wofv, prts);

    reduce_heads_kernel<<<(BB * SS + 255) / 256, 256, 0, stream>>>(
        prts, wofv, out);
}
